// top_k_gating_15573551415342
// MI455X (gfx1250) — matrix operand order repaired and checked by reading, not compiled, not run on hardware
//
#include <hip/hip_runtime.h>
#include <hip/hip_bf16.h>
#include <cstdint>
#include <cstddef>

#define N_TOK 65536
#define DIM   2048
#define NEXP  64
#define TOPK  8

typedef __attribute__((ext_vector_type(8)))  float          v8f;
typedef __attribute__((ext_vector_type(16))) unsigned short v16u;
typedef __attribute__((ext_vector_type(16))) __bf16         v16bf;

// ---------------- accumulator init (d_ws is poisoned, must self-init) ------
__global__ void zero_accums_kernel(float* __restrict__ acc) {
  int i = threadIdx.x;
  if (i < 129) acc[i] = 0.0f;  // ps[64], fr[64], z[1]
}

// ---------------- W pre-swizzle: f32 [E,D] -> bf16 hi/lo B-fragments -------
// Fragment index layout: frag = (c*4 + t)*32 + lane, 16 bf16 per lane.
// B 16-bit 32x16 layout: lanes 0-15 hold K=0..7,16..23, lanes 16-31 hold K=8..15,24..31,
// N (expert) = lane%16; element slot s within lane <-> K = 8*(lane/16)+(s&7)+16*(s>>3).
__global__ void swizzle_w_kernel(const float* __restrict__ W,
                                 unsigned short* __restrict__ Whi,
                                 unsigned short* __restrict__ Wlo) {
  int i    = blockIdx.x * 256 + threadIdx.x;  // 0 .. E*D-1 = 131071
  int j    = i & 15;
  int lane = (i >> 4) & 31;
  int t    = (i >> 9) & 3;
  int c    = i >> 11;
  int e = t * 16 + (lane & 15);
  int k = c * 32 + (lane >> 4) * 8 + (j & 7) + (j >> 3) * 16;
  float w = W[(size_t)e * DIM + k];
  __bf16 h = (__bf16)w;                    // native v_cvt, RNE
  __bf16 l = (__bf16)(w - (float)h);       // exact residual -> bf16
  Whi[i] = __builtin_bit_cast(unsigned short, h);
  Wlo[i] = __builtin_bit_cast(unsigned short, l);
}

// ---------------- main: GEMM (split-bf16 WMMA) + top-k + aux stats ---------
// Each wave owns 32 rows (two 16-row M-tiles) x all 64 experts:
// 8 f32 accumulators, 24 WMMAs per K-32 chunk, B fragments reused 6x each.
__global__ __launch_bounds__(128) void router_kernel(
    const float* __restrict__ x,
    const unsigned short* __restrict__ Whi,
    const unsigned short* __restrict__ Wlo,
    int* __restrict__ out_idx, float* __restrict__ out_gate,
    float* __restrict__ g_ps, float* __restrict__ g_fr,
    float* __restrict__ g_z) {
  __shared__ float lg[128][65];  // 128 rows x 64 logits, padded stride
  __shared__ float s_ps[64];
  __shared__ float s_fr[64];
  __shared__ float s_z;

  const int tid  = threadIdx.x;
  const int wave = tid >> 5;
  const int lane = tid & 31;
  const int m    = lane & 15;   // A row within tile (lanes 16-31 repeat rows)
  const int half = lane >> 4;   // selects complementary K half

  if (tid < 64) { s_ps[tid] = 0.0f; s_fr[tid] = 0.0f; }
  if (tid == 0) s_z = 0.0f;

  const int rowBase = blockIdx.x * 128 + wave * 32;
  const float* xr[2];
  xr[0] = x + (size_t)(rowBase + m) * DIM;
  xr[1] = x + (size_t)(rowBase + 16 + m) * DIM;
  const int k0 = half * 8;  // lane's first K-run; second run at +16

  v8f acc[2][4];
  #pragma unroll
  for (int mt = 0; mt < 2; ++mt)
    #pragma unroll
    for (int t = 0; t < 4; ++t) acc[mt][t] = v8f{};

  // software pipeline: xn holds chunk c's x data at the top of iteration c;
  // we convert it to A fragments first, then refill xn with chunk c+1.
  float4 xn[2][4];
  #pragma unroll
  for (int mt = 0; mt < 2; ++mt) {
    xn[mt][0] = *(const float4*)(xr[mt] + k0);
    xn[mt][1] = *(const float4*)(xr[mt] + k0 + 4);
    xn[mt][2] = *(const float4*)(xr[mt] + k0 + 16);
    xn[mt][3] = *(const float4*)(xr[mt] + k0 + 20);
  }

  #pragma unroll 1
  for (int c = 0; c < DIM / 32; ++c) {
    // native bf16 split of chunk c: x = hi + lo (consumes xn, no copies)
    v16bf Ahi[2], Alo[2];
    #pragma unroll
    for (int mt = 0; mt < 2; ++mt) {
      const float fxv[16] = {xn[mt][0].x, xn[mt][0].y, xn[mt][0].z, xn[mt][0].w,
                             xn[mt][1].x, xn[mt][1].y, xn[mt][1].z, xn[mt][1].w,
                             xn[mt][2].x, xn[mt][2].y, xn[mt][2].z, xn[mt][2].w,
                             xn[mt][3].x, xn[mt][3].y, xn[mt][3].z, xn[mt][3].w};
      #pragma unroll
      for (int s = 0; s < 16; ++s) {
        const __bf16 h = (__bf16)fxv[s];
        Ahi[mt][s] = h;
        Alo[mt][s] = (__bf16)(fxv[s] - (float)h);
      }
    }

    // refill xn with chunk c+1 (issues early, lands during the WMMA block)
    if (c < DIM / 32 - 1) {
      const int kb = (c + 1) * 32 + k0;
      #pragma unroll
      for (int mt = 0; mt < 2; ++mt) {
        xn[mt][0] = *(const float4*)(xr[mt] + kb);
        xn[mt][1] = *(const float4*)(xr[mt] + kb + 4);
        xn[mt][2] = *(const float4*)(xr[mt] + kb + 16);
        xn[mt][3] = *(const float4*)(xr[mt] + kb + 20);
        __builtin_prefetch(xr[mt] + kb + 256, 0, 1);  // ~1KB ahead on stream
      }
    }

    const v16u* bhp = (const v16u*)Whi + ((size_t)c * 128 + lane);
    const v16u* blp = (const v16u*)Wlo + ((size_t)c * 128 + lane);
    __builtin_prefetch(bhp + 128, 0, 1);  // warm next chunk's B fragments
    __builtin_prefetch(blp + 128, 0, 1);

    #pragma unroll
    for (int t = 0; t < 4; ++t) {
      const v16bf Bh = __builtin_bit_cast(v16bf, bhp[t * 32]);
      const v16bf Bl = __builtin_bit_cast(v16bf, blp[t * 32]);
      #pragma unroll
      for (int mt = 0; mt < 2; ++mt) {
        // logits += xhi*whi + xlo*whi + xhi*wlo  (~fp32 accuracy)
        acc[mt][t] = __builtin_amdgcn_wmma_f32_16x16x32_bf16(
            false, Ahi[mt], false, Bh, (short)0, acc[mt][t], false, false);
        acc[mt][t] = __builtin_amdgcn_wmma_f32_16x16x32_bf16(
            false, Alo[mt], false, Bh, (short)0, acc[mt][t], false, false);
        acc[mt][t] = __builtin_amdgcn_wmma_f32_16x16x32_bf16(
            false, Ahi[mt], false, Bl, (short)0, acc[mt][t], false, false);
      }
    }
  }

  // C/D layout: vgpr r, lanes 0-15 -> M=r, lanes 16-31 -> M=8+r; N=lane%16
  #pragma unroll
  for (int mt = 0; mt < 2; ++mt)
    #pragma unroll
    for (int t = 0; t < 4; ++t)
      #pragma unroll
      for (int r = 0; r < 8; ++r)
        lg[wave * 32 + mt * 16 + half * 8 + r][t * 16 + m] = acc[mt][t][r];

  __syncthreads();

  // -------- per-row phase: one thread per row (128 rows / block) ----------
  {
    const int grow = blockIdx.x * 128 + tid;
    float mx = -3.402823466e38f;
    #pragma unroll 8
    for (int e = 0; e < 64; ++e) mx = fmaxf(mx, lg[tid][e]);
    float denom = 0.0f;
    #pragma unroll 8
    for (int e = 0; e < 64; ++e) denom += expf(lg[tid][e] - mx);
    const float inv = 1.0f / denom;
    #pragma unroll 4
    for (int e = 0; e < 64; ++e)
      atomicAdd(&s_ps[e], expf(lg[tid][e] - mx) * inv);

    // zloss term: logaddexp(lse, log(EPS))^2, stable
    const float lse = mx + logf(denom);
    const float logeps = -11.512925464970229f;  // log(1e-5)
    const float a = fmaxf(lse, logeps), b = fminf(lse, logeps);
    const float l2 = a + log1pf(expf(b - a));
    atomicAdd(&s_z, l2 * l2);

    // top-8 by iterative max; strict '>' ties to lowest index (matches top_k)
    unsigned long long used = 0ull;
    float tv[TOPK]; int ti[TOPK];
    #pragma unroll
    for (int k = 0; k < TOPK; ++k) {
      float best = -3.402823466e38f; int bi = 0;
      #pragma unroll 1
      for (int e = 0; e < 64; ++e) {
        const float v = lg[tid][e];
        if ((((used >> e) & 1ull) == 0ull) && v > best) { best = v; bi = e; }
      }
      used |= (1ull << bi);
      tv[k] = best; ti[k] = bi;
    }
    float ge[TOPK]; float gs = 0.0f;
    #pragma unroll
    for (int k = 0; k < TOPK; ++k) { ge[k] = expf(tv[k] - tv[0]); gs += ge[k]; }
    const float gi = 1.0f / gs;
    #pragma unroll
    for (int k = 0; k < TOPK; ++k) {
      out_idx[(size_t)grow * TOPK + k]  = ti[k];
      out_gate[(size_t)grow * TOPK + k] = ge[k] * gi;
      atomicAdd(&s_fr[ti[k]], 1.0f);
    }
  }
  __syncthreads();

  if (tid < 64) {
    atomicAdd(&g_ps[tid], s_ps[tid]);
    atomicAdd(&g_fr[tid], s_fr[tid]);
  }
  if (tid == 0) atomicAdd(g_z, s_z);
}

// ---------------- scalar loss --------------------------------------------
__global__ void finalize_kernel(const float* __restrict__ g_ps,
                                const float* __restrict__ g_fr,
                                const float* __restrict__ g_z,
                                float* __restrict__ out_loss) {
  if (threadIdx.x == 0) {
    float s1 = 0.0f, s2 = 0.0f;
    for (int e = 0; e < 64; ++e) { s1 += fabsf(g_ps[e]); s2 += fabsf(g_fr[e]); }
    s1 = fmaxf(s1, 1e-12f);
    s2 = fmaxf(s2, 1e-12f);
    float dot = 0.0f;
    for (int e = 0; e < 64; ++e) dot += (g_ps[e] / s1) * (g_fr[e] / s2);
    out_loss[0] = (float)NEXP * dot + 0.1f * (g_z[0] / (float)N_TOK);
  }
}

extern "C" void kernel_launch(void* const* d_in, const int* in_sizes, int n_in,
                              void* d_out, int out_size, void* d_ws, size_t ws_size,
                              hipStream_t stream) {
  const float* x = (const float*)d_in[0];   // [N, D] f32
  const float* W = (const float*)d_in[1];   // [E, D] f32

  // workspace: Whi (256KB) | Wlo (256KB) | accums (129 f32)
  unsigned short* Whi = (unsigned short*)d_ws;
  unsigned short* Wlo = Whi + (size_t)NEXP * DIM;
  float* accums = (float*)(Wlo + (size_t)NEXP * DIM);
  float* g_ps = accums;
  float* g_fr = accums + 64;
  float* g_z  = accums + 128;

  // output: [top_i int32 N*K][gates f32 N*K][loss f32]
  int*   out_idx  = (int*)d_out;
  float* out_gate = (float*)d_out + (size_t)N_TOK * TOPK;
  float* out_loss = (float*)d_out + (size_t)2 * N_TOK * TOPK;

  zero_accums_kernel<<<1, 256, 0, stream>>>(accums);
  swizzle_w_kernel<<<(NEXP * DIM) / 256, 256, 0, stream>>>(W, Whi, Wlo);
  router_kernel<<<N_TOK / 128, 128, 0, stream>>>(x, Whi, Wlo, out_idx, out_gate,
                                                 g_ps, g_fr, g_z);
  finalize_kernel<<<1, 64, 0, stream>>>(g_ps, g_fr, g_z, out_loss);
}
